// DocREModel_50276887167443
// MI455X (gfx1250) — matrix operand order repaired and checked by reading, not compiled, not run on hardware
//
#include <hip/hip_runtime.h>
#include <hip/hip_bf16.h>
#include <math.h>

// ---------------------------------------------------------------- dimensions
#define N_B     4
#define C_SEQ   1024
#define D_HID   768
#define H_HEADS 12
#define E_ENT   30
#define M_MEN   4
#define P_PAIR  870
#define EMB     768
#define L_LAB   97
#define NROWS   (N_B * P_PAIR)     /* 3480 */
#define FEAT    (2 * D_HID)        /* 1536 */
#define KBIL    (EMB * 64)         /* 49152 */
#define LTILES  7                  /* ceil(97/16) */
#define LPAD    (LTILES * 16)      /* 112 */
#define KC      128                /* bilinear LDS K-chunk */

typedef __bf16 bf16_t;
typedef bf16_t bfrag_t __attribute__((ext_vector_type(16)));
typedef float  v8f     __attribute__((ext_vector_type(8)));
typedef int    v4i_t   __attribute__((vector_size(16)));   // matches builtin

typedef __attribute__((address_space(1))) v4i_t* g_v4i_ptr;
typedef __attribute__((address_space(3))) v4i_t* l_v4i_ptr;

// ---- CDNA5 async global->LDS path (probe which builtins this clang has)
#if defined(__has_builtin)
#  if __has_builtin(__builtin_amdgcn_global_load_async_to_lds_b128)
#    define HAVE_ASYNC_LDS 1
#    pragma message("CDNA5: using __builtin_amdgcn_global_load_async_to_lds_b128")
#  endif
#  if __has_builtin(__builtin_amdgcn_s_wait_asynccnt)
#    define HAVE_WAIT_ASYNC_BUILTIN 1
#    pragma message("CDNA5: using __builtin_amdgcn_s_wait_asynccnt")
#  endif
#endif
#ifndef HAVE_ASYNC_LDS
#  define HAVE_ASYNC_LDS 0
#  pragma message("CDNA5: async-to-LDS builtin NOT available, ds_store fallback")
#endif
#ifndef HAVE_WAIT_ASYNC_BUILTIN
#  define HAVE_WAIT_ASYNC_BUILTIN 0
#endif

__device__ __forceinline__ void wait_asynccnt0() {
#if HAVE_WAIT_ASYNC_BUILTIN
    __builtin_amdgcn_s_wait_asynccnt(0);
#elif HAVE_ASYNC_LDS
    asm volatile("s_wait_asynccnt 0x0" ::: "memory");
#endif
}

__device__ __forceinline__ int imin(int a, int b) { return a < b ? a : b; }

// Pack two 8x-bf16 128-bit loads into one 16x-bf16 fragment.
__device__ __forceinline__ bfrag_t make_frag(uint4 u0, uint4 u1) {
    union { uint4 u; bf16_t h[8]; } a, b;
    a.u = u0; b.u = u1;
    bfrag_t f;
#pragma unroll
    for (int t = 0; t < 8; ++t) { f[t] = a.h[t]; f[8 + t] = b.h[t]; }
    return f;
}

__device__ __forceinline__ v8f wmma_bf16(const bfrag_t& a, const bfrag_t& b,
                                         const v8f& c) {
    return __builtin_amdgcn_wmma_f32_16x16x32_bf16(
        /*neg_a=*/false, a, /*neg_b=*/false, b,
        /*c_mod=*/(short)0, c, /*reuse_a=*/false, /*reuse_b=*/false);
}

// ----------------------------------------------------- elementwise transposes
__global__ void seq_transpose_kernel(const float* __restrict__ seq,
                                     bf16_t* __restrict__ seqT) {
    const size_t total = (size_t)N_B * C_SEQ * D_HID;
    for (size_t i = (size_t)blockIdx.x * blockDim.x + threadIdx.x; i < total;
         i += (size_t)gridDim.x * blockDim.x) {
        const size_t nb  = i / ((size_t)C_SEQ * D_HID);
        const size_t rem = i % ((size_t)C_SEQ * D_HID);
        const size_t dd  = rem / C_SEQ;
        const size_t cc  = rem % C_SEQ;
        seqT[i] = (bf16_t)seq[(nb * C_SEQ + cc) * D_HID + dd];
    }
}

// out [C x R] bf16 = transpose(in [R x C] f32)
__global__ void transpose_bf16_kernel(const float* __restrict__ in,
                                      bf16_t* __restrict__ out,
                                      int R, int C) {
    const size_t total = (size_t)R * C;
    for (size_t i = (size_t)blockIdx.x * blockDim.x + threadIdx.x; i < total;
         i += (size_t)gridDim.x * blockDim.x) {
        const size_t c = i / R;
        const size_t r = i % R;
        out[i] = (bf16_t)in[r * (size_t)C + c];
    }
}

// ------------------------------------------------- entity pooling (logsumexp)
__global__ void entity_emb_kernel(const float* __restrict__ seq,
                                  const int* __restrict__ midx,
                                  const unsigned char* __restrict__ mmask,
                                  float* __restrict__ e_emb) {
    const int be = blockIdx.x;             // n*E
    const int nb = be / E_ENT, e = be % E_ENT;
    int idx[M_MEN]; int valid[M_MEN];
#pragma unroll
    for (int m = 0; m < M_MEN; ++m) {
        idx[m]   = midx[(be)*M_MEN + m];
        valid[m] = mmask[(be)*M_MEN + m] != 0;
    }
    for (int d0 = threadIdx.x; d0 < D_HID; d0 += blockDim.x) {
        float mx = -INFINITY;
#pragma unroll
        for (int m = 0; m < M_MEN; ++m)
            if (valid[m])
                mx = fmaxf(mx, seq[((size_t)nb * C_SEQ + idx[m]) * D_HID + d0]);
        float s = 0.f;
#pragma unroll
        for (int m = 0; m < M_MEN; ++m)
            if (valid[m])
                s += expf(seq[((size_t)nb * C_SEQ + idx[m]) * D_HID + d0] - mx);
        const float r = (s > 0.f) ? (mx + logf(s)) : -INFINITY;
        e_emb[((size_t)nb * E_ENT + e) * D_HID + d0] = r;
    }
}

// ------------------------------------------- entity attention (mention mean)
__global__ void entity_att_kernel(const float* __restrict__ att,
                                  const int* __restrict__ midx,
                                  const unsigned char* __restrict__ mmask,
                                  float* __restrict__ e_att) {
    const int id  = blockIdx.x;            // n*E*h
    const int nb  = id / (E_ENT * H_HEADS);
    const int rem = id % (E_ENT * H_HEADS);
    const int e   = rem / H_HEADS, hh = rem % H_HEADS;
    int idx[M_MEN]; int valid[M_MEN]; float cnt = 0.f;
#pragma unroll
    for (int m = 0; m < M_MEN; ++m) {
        idx[m]   = midx[(nb * E_ENT + e) * M_MEN + m];
        valid[m] = mmask[(nb * E_ENT + e) * M_MEN + m] != 0;
        cnt += valid[m] ? 1.f : 0.f;
    }
    const float inv = 1.f / fmaxf(cnt, 1.f);
    const float* __restrict__ base =
        att + ((size_t)nb * H_HEADS + hh) * C_SEQ * C_SEQ;
    float* __restrict__ out =
        e_att + (((size_t)nb * E_ENT + e) * H_HEADS + hh) * C_SEQ;
    for (int c0 = threadIdx.x; c0 < C_SEQ; c0 += blockDim.x) {
        float s = 0.f;
#pragma unroll
        for (int m = 0; m < M_MEN; ++m)
            if (valid[m]) s += base[(size_t)idx[m] * C_SEQ + c0];
        out[c0] = s * inv;
    }
}

// ------------------------------- pair attention (mean over heads + normalize)
__global__ void pair_att_kernel(const float* __restrict__ e_att,
                                const int* __restrict__ hts,
                                bf16_t* __restrict__ htA) {
    const int id = blockIdx.x;             // n*P
    const int nb = id / P_PAIR;
    const int hi = hts[(size_t)id * 2 + 0];
    const int ti = hts[(size_t)id * 2 + 1];
    const float* __restrict__ ha =
        e_att + ((size_t)nb * E_ENT + hi) * H_HEADS * C_SEQ;
    const float* __restrict__ ta =
        e_att + ((size_t)nb * E_ENT + ti) * H_HEADS * C_SEQ;

    float vals[4]; float local = 0.f;
#pragma unroll
    for (int q = 0; q < 4; ++q) {
        const int c0 = threadIdx.x + q * 256;
        float s = 0.f;
#pragma unroll
        for (int hh = 0; hh < H_HEADS; ++hh)
            s += ha[(size_t)hh * C_SEQ + c0] * ta[(size_t)hh * C_SEQ + c0];
        s *= (1.f / H_HEADS);
        vals[q] = s; local += s;
    }
    __shared__ float red[256];
    red[threadIdx.x] = local;
    __syncthreads();
    for (int off = 128; off > 0; off >>= 1) {
        if (threadIdx.x < off) red[threadIdx.x] += red[threadIdx.x + off];
        __syncthreads();
    }
    const float scale = 1.f / (red[0] + 1e-5f);
#pragma unroll
    for (int q = 0; q < 4; ++q)
        htA[(size_t)id * C_SEQ + threadIdx.x + q * 256] =
            (bf16_t)(vals[q] * scale);
}

// ---------------------------------------- rs = ht_att @ seq  (batched, WMMA)
__global__ void rs_gemm_kernel(const bf16_t* __restrict__ htA,
                               const bf16_t* __restrict__ seqT,
                               float* __restrict__ rs) {
    const int tm = blockIdx.x, tn = blockIdx.y, nb = blockIdx.z;
    const int lane = threadIdx.x & 31;
    const int m = lane & 15, kh = lane >> 4, nc = lane & 15;
    const bf16_t* __restrict__ A  = htA  + (size_t)nb * P_PAIR * C_SEQ;
    const bf16_t* __restrict__ BT = seqT + (size_t)nb * D_HID * C_SEQ;

    // clamped row (stores are guarded; clamped rows feed unstored outputs)
    const int rowA = imin(tm * 16 + m, P_PAIR - 1);
    const bf16_t* pa = A  + (size_t)rowA * C_SEQ + kh * 8;
    const bf16_t* pb = BT + (size_t)(tn * 16 + nc) * C_SEQ + kh * 8;

    v8f acc = {0.f, 0.f, 0.f, 0.f, 0.f, 0.f, 0.f, 0.f};
    for (int k0 = 0; k0 < C_SEQ; k0 += 32) {
        __builtin_prefetch(pb + 64, 0, 1);
        bfrag_t a = make_frag(*(const uint4*)pa, *(const uint4*)(pa + 16));
        bfrag_t b = make_frag(*(const uint4*)pb, *(const uint4*)(pb + 16));
        acc = wmma_bf16(a, b, acc);
        pa += 32; pb += 32;
    }
    float* __restrict__ C = rs + (size_t)nb * P_PAIR * D_HID;
    const int half  = lane >> 4;
    const int rbase = tm * 16 + half * 8;
    float* cp = C + (size_t)rbase * D_HID + tn * 16 + nc;
    if (rbase + 8 <= P_PAIR) {               // fast path: whole strip in-range
#pragma unroll
        for (int r = 0; r < 8; ++r) cp[(size_t)r * D_HID] = acc[r];
    } else {
#pragma unroll
        for (int r = 0; r < 8; ++r)
            if (rbase + r < P_PAIR) cp[(size_t)r * D_HID] = acc[r];
    }
}

// -------------------------------------------- pair features [e_emb | rs] bf16
__global__ void feat_kernel(const float* __restrict__ e_emb,
                            const float* __restrict__ rs,
                            const int* __restrict__ hts,
                            bf16_t* __restrict__ featH,
                            bf16_t* __restrict__ featT) {
    const int row = blockIdx.x;            // NROWS
    const int nb  = row / P_PAIR;
    const int hi  = hts[(size_t)row * 2 + 0];
    const int ti  = hts[(size_t)row * 2 + 1];
    const float* __restrict__ eh = e_emb + ((size_t)nb * E_ENT + hi) * D_HID;
    const float* __restrict__ et = e_emb + ((size_t)nb * E_ENT + ti) * D_HID;
    const float* __restrict__ rr = rs + (size_t)row * D_HID;
    for (int q = threadIdx.x; q < FEAT; q += blockDim.x) {
        if (q < D_HID) {
            featH[(size_t)row * FEAT + q] = (bf16_t)eh[q];
            featT[(size_t)row * FEAT + q] = (bf16_t)et[q];
        } else {
            const bf16_t v = (bf16_t)rr[q - D_HID];
            featH[(size_t)row * FEAT + q] = v;
            featT[(size_t)row * FEAT + q] = v;
        }
    }
}

// ------------------------- hs/ts = tanh(feat @ W + b)   (WMMA + f32 epilogue)
__global__ void proj_gemm_kernel(const bf16_t* __restrict__ featH,
                                 const bf16_t* __restrict__ featT,
                                 const bf16_t* __restrict__ WhT,
                                 const bf16_t* __restrict__ WtT,
                                 const float* __restrict__ bh,
                                 const float* __restrict__ bt,
                                 bf16_t* __restrict__ hs,
                                 bf16_t* __restrict__ ts) {
    const int tm = blockIdx.x, tn = blockIdx.y, which = blockIdx.z;
    const int lane = threadIdx.x & 31;
    const int m = lane & 15, kh = lane >> 4, nc = lane & 15;
    const bf16_t* __restrict__ A    = which ? featT : featH;
    const bf16_t* __restrict__ BT   = which ? WtT   : WhT;
    const float*  __restrict__ bias = which ? bt    : bh;
    bf16_t*       __restrict__ Out  = which ? ts    : hs;

    const int rowA = imin(tm * 16 + m, NROWS - 1);
    const bf16_t* pa = A  + (size_t)rowA * FEAT + kh * 8;
    const bf16_t* pb = BT + (size_t)(tn * 16 + nc) * FEAT + kh * 8;

    v8f acc = {0.f, 0.f, 0.f, 0.f, 0.f, 0.f, 0.f, 0.f};
    for (int k0 = 0; k0 < FEAT; k0 += 32) {
        __builtin_prefetch(pb + 64, 0, 1);
        bfrag_t a = make_frag(*(const uint4*)pa, *(const uint4*)(pa + 16));
        bfrag_t b = make_frag(*(const uint4*)pb, *(const uint4*)(pb + 16));
        acc = wmma_bf16(a, b, acc);
        pa += 32; pb += 32;
    }
    const int half  = lane >> 4;
    const int col   = tn * 16 + nc;
    const float bv  = bias[col];
    const int rbase = tm * 16 + half * 8;
    bf16_t* op = Out + (size_t)rbase * EMB + col;
    if (rbase + 8 <= NROWS) {
#pragma unroll
        for (int r = 0; r < 8; ++r)
            op[(size_t)r * EMB] = (bf16_t)tanhf(acc[r] + bv);
    } else {
#pragma unroll
        for (int r = 0; r < 8; ++r)
            if (rbase + r < NROWS)
                op[(size_t)r * EMB] = (bf16_t)tanhf(acc[r] + bv);
    }
}

// ---------------- fused grouped-bilinear classifier (on-the-fly outer product)
// 8 waves / block share a 112 x KC bf16 slice of W_bil^T staged in LDS.
// logits[row,l] = sum_k hs[row,kblk*64+i]*ts[row,kblk*64+j]*W_bil[k,l] + b[l]
__global__ __launch_bounds__(256) void
bilinear_kernel(const bf16_t* __restrict__ hs, const bf16_t* __restrict__ ts,
                const bf16_t* __restrict__ WbT, const float* __restrict__ b_bil,
                float* __restrict__ out) {
    __shared__ __align__(16) bf16_t sW[LPAD * KC];   // 28 KB

    const int tid  = threadIdx.x;
    const int lane = tid & 31;
    const int wave = tid >> 5;
    const int m = lane & 15, kh = lane >> 4, nc = lane & 15;

    const int row0 = blockIdx.x * 128 + wave * 16;
    const int row  = imin(row0 + m, NROWS - 1);      // clamped A row
    const bf16_t* __restrict__ hrow = hs + (size_t)row * EMB;
    const bf16_t* __restrict__ trow = ts + (size_t)row * EMB;

    v8f acc[LTILES];
#pragma unroll
    for (int lt = 0; lt < LTILES; ++lt)
#pragma unroll
        for (int r = 0; r < 8; ++r) acc[lt][r] = 0.f;

    for (int kk0 = 0; kk0 < KBIL; kk0 += KC) {
        // ---- cooperative stage of WbT[0..111, kk0..kk0+KC) into LDS
        // 112*KC/8 = 1792 16B chunks, 7 per thread
#pragma unroll
        for (int q = 0; q < 7; ++q) {
            const int ci = tid + q * 256;
            const int l  = ci >> 4;                  // KC/8 = 16 chunks per l
            const int c8 = ci & 15;
            const int lc = imin(l, L_LAB - 1);       // clamp OOB label rows
            const bf16_t* gp = WbT + (size_t)lc * KBIL + kk0 + c8 * 8;
            bf16_t*       lp = sW + l * KC + c8 * 8;
#if HAVE_ASYNC_LDS
            __builtin_amdgcn_global_load_async_to_lds_b128(
                (g_v4i_ptr)(uintptr_t)gp,            // v4i addrspace(1)*
                (l_v4i_ptr)lp,                       // v4i addrspace(3)*
                /*offset=*/0, /*cpol=*/0);
#else
            *(uint4*)lp = *(const uint4*)gp;
#endif
        }
        __builtin_prefetch(WbT + (size_t)(tid >> 4) * KBIL + kk0 + KC, 0, 1);
        wait_asynccnt0();
        __syncthreads();

        // ---- compute on the staged chunk
#pragma unroll
        for (int ks = 0; ks < KC; ks += 32) {
            const int kk   = kk0 + ks;
            const int kblk = kk >> 12;
            const int iIdx = (kk >> 6) & 63;
            const int j0   = kk & 63;                // 0 or 32

            // A fragment: bl[row, kk..kk+31] = hs[row,·i] * ts[row,·j]
            const float hsv = (float)hrow[kblk * 64 + iIdx];
            const bf16_t* tp = trow + kblk * 64 + j0 + kh * 8;
            union { uint4 u; bf16_t h[8]; } t0, t1;
            t0.u = *(const uint4*)(tp);
            t1.u = *(const uint4*)(tp + 16);
            bfrag_t a;
#pragma unroll
            for (int t = 0; t < 8; ++t) {
                a[t]     = (bf16_t)(hsv * (float)t0.h[t]);
                a[8 + t] = (bf16_t)(hsv * (float)t1.h[t]);
            }

#pragma unroll
            for (int lt = 0; lt < LTILES; ++lt) {
                const bf16_t* sp = sW + (lt * 16 + nc) * KC + ks + kh * 8;
                bfrag_t b = make_frag(*(const uint4*)sp, *(const uint4*)(sp + 16));
                acc[lt] = wmma_bf16(a, b, acc[lt]);
            }
        }
        __syncthreads();
    }

    // ---- epilogue
    const int half  = lane >> 4;
    const int rbase = row0 + half * 8;
    const bool fullrows = (rbase + 8 <= NROWS);
#pragma unroll
    for (int lt = 0; lt < LTILES; ++lt) {
        const int l = lt * 16 + nc;
        if (l >= L_LAB) continue;
        const float bv = b_bil[l];
        float* op = out + (size_t)rbase * L_LAB + l;
        if (fullrows) {
#pragma unroll
            for (int r = 0; r < 8; ++r) op[(size_t)r * L_LAB] = acc[lt][r] + bv;
        } else {
#pragma unroll
            for (int r = 0; r < 8; ++r)
                if (rbase + r < NROWS) op[(size_t)r * L_LAB] = acc[lt][r] + bv;
        }
    }
}

// ---------------------------------------------------------------------- host
extern "C" void kernel_launch(void* const* d_in, const int* in_sizes, int n_in,
                              void* d_out, int out_size, void* d_ws,
                              size_t ws_size, hipStream_t stream) {
    (void)in_sizes; (void)n_in; (void)out_size; (void)ws_size;

    const float*         seq   = (const float*)d_in[0];
    const float*         att   = (const float*)d_in[1];
    const int*           midx  = (const int*)d_in[2];
    const unsigned char* mmask = (const unsigned char*)d_in[3];
    const int*           hts   = (const int*)d_in[4];
    const float*         Wh    = (const float*)d_in[5];
    const float*         bh    = (const float*)d_in[6];
    const float*         Wt    = (const float*)d_in[7];
    const float*         bt    = (const float*)d_in[8];
    const float*         Wb    = (const float*)d_in[9];
    const float*         bb    = (const float*)d_in[10];
    float*               out   = (float*)d_out;

    // ---- workspace carve-out (256B aligned)
    char* w = (char*)d_ws;
    size_t off = 0;
    auto alloc = [&](size_t bytes) -> void* {
        void* p = w + off;
        off = (off + bytes + 255) & ~(size_t)255;
        return p;
    };
    bf16_t* seqT  = (bf16_t*)alloc((size_t)N_B * D_HID * C_SEQ * 2);
    bf16_t* WhT   = (bf16_t*)alloc((size_t)EMB * FEAT * 2);
    bf16_t* WtT   = (bf16_t*)alloc((size_t)EMB * FEAT * 2);
    bf16_t* WbT   = (bf16_t*)alloc((size_t)L_LAB * KBIL * 2);
    float*  e_emb = (float*) alloc((size_t)N_B * E_ENT * D_HID * 4);
    float*  e_att = (float*) alloc((size_t)N_B * E_ENT * H_HEADS * C_SEQ * 4);
    bf16_t* htA   = (bf16_t*)alloc((size_t)N_B * P_PAIR * C_SEQ * 2);
    float*  rsbuf = (float*) alloc((size_t)NROWS * D_HID * 4);
    bf16_t* featH = (bf16_t*)alloc((size_t)NROWS * FEAT * 2);
    bf16_t* featT = (bf16_t*)alloc((size_t)NROWS * FEAT * 2);
    bf16_t* hsbuf = (bf16_t*)alloc((size_t)NROWS * EMB * 2);
    bf16_t* tsbuf = (bf16_t*)alloc((size_t)NROWS * EMB * 2);

    // 1) bf16 transposes
    seq_transpose_kernel<<<2048, 256, 0, stream>>>(seq, seqT);
    transpose_bf16_kernel<<<1024, 256, 0, stream>>>(Wh, WhT, FEAT, EMB);
    transpose_bf16_kernel<<<1024, 256, 0, stream>>>(Wt, WtT, FEAT, EMB);
    transpose_bf16_kernel<<<4096, 256, 0, stream>>>(Wb, WbT, KBIL, L_LAB);

    // 2) entity pooling
    entity_emb_kernel<<<N_B * E_ENT, 256, 0, stream>>>(seq, midx, mmask, e_emb);
    entity_att_kernel<<<N_B * E_ENT * H_HEADS, 256, 0, stream>>>(att, midx,
                                                                 mmask, e_att);

    // 3) pair attention (bf16 GEMM-A)
    pair_att_kernel<<<N_B * P_PAIR, 256, 0, stream>>>(e_att, hts, htA);

    // 4) rs GEMM
    {
        dim3 grid((P_PAIR + 15) / 16, D_HID / 16, N_B);
        rs_gemm_kernel<<<grid, 32, 0, stream>>>(htA, seqT, rsbuf);
    }

    // 5) pair features
    feat_kernel<<<NROWS, 256, 0, stream>>>(e_emb, rsbuf, hts, featH, featT);

    // 6) projections
    {
        dim3 grid((NROWS + 15) / 16, EMB / 16, 2);
        proj_gemm_kernel<<<grid, 32, 0, stream>>>(featH, featT, WhT, WtT, bh,
                                                  bt, hsbuf, tsbuf);
    }

    // 7) fused bilinear classifier (8 waves/block, async-LDS-staged W tile)
    bilinear_kernel<<<(NROWS + 127) / 128, 256, 0, stream>>>(hsbuf, tsbuf, WbT,
                                                             bb, out);
}
